// MultiheadFlashlinearrope_55018531062058
// MI455X (gfx1250) — matrix operand order repaired and checked by reading, not compiled, not run on hardware
//
#include <hip/hip_runtime.h>

typedef __bf16 bf16;
typedef __attribute__((ext_vector_type(16))) __bf16 v16bf;
typedef __attribute__((ext_vector_type(8)))  __bf16 v8bf;
typedef __attribute__((ext_vector_type(8)))  float  v8f;
typedef __attribute__((ext_vector_type(4))) unsigned int u32x4;
typedef __attribute__((ext_vector_type(8))) int i32x8;
typedef __attribute__((ext_vector_type(4))) int i32x4;

#define BB 2
#define SS 8192
#define EE 1024
#define HH 16
#define DD 64
#define CC 128
#define NCC 64  // SS/CC

// ---------------- WMMA helpers (CDNA5 bf16, f32 accum) ----------------
__device__ __forceinline__ v8f wmma_bf16(v16bf a, v16bf b, v8f c) {
  return __builtin_amdgcn_wmma_f32_16x16x32_bf16(false, a, false, b, (short)0, c,
                                                 false, false);
}

// A fragment (16x32): lane holds row m0+(lane&15); K chunk [kh,kh+16)
__device__ __forceinline__ v16bf frag_a(const bf16* mat, int ld, int m0, int kb, int lane) {
  int m  = m0 + (lane & 15);
  int kh = (lane >> 4) * 16;
  const bf16* p = mat + (size_t)m * ld + kb;
  v8bf lo = *(const v8bf*)(p + kh);
  v8bf hi = *(const v8bf*)(p + kh + 8);
  v16bf a;
#pragma unroll
  for (int i = 0; i < 8; ++i) { a[i] = lo[i]; a[i + 8] = hi[i]; }
  return a;
}

// B fragment (32x16) from matrix stored as B^T rows (matT[n][k])
__device__ __forceinline__ v16bf frag_b(const bf16* matT, int ld, int n0, int kb, int lane) {
  int n  = n0 + (lane & 15);
  int kh = (lane >> 4) * 16;
  return *(const v16bf*)(matT + (size_t)n * ld + kb + kh);
}

// ---------------- TDM: async 2D bf16 tile load, global -> LDS ----------------
// Builds D# groups per CDNA5 ISA 8.3/8.4 and issues TENSOR_LOAD_TO_LDS.
// All dims in bf16 elements. Completion via s_wait_tensorcnt.
__device__ __forceinline__ void tdm_load_2d(unsigned lds_off, const void* gaddr,
                                            unsigned tensor_w, unsigned tensor_h,
                                            unsigned tile_w, unsigned tile_h,
                                            unsigned stride_elems) {
  unsigned long long ga = (unsigned long long)(size_t)gaddr;
  u32x4 g0;
  g0[0] = 1u;                                       // count=1, user descriptor
  g0[1] = lds_off;                                  // lds_addr [63:32]
  g0[2] = (unsigned)(ga & 0xFFFFFFFFu);             // global_addr [95:64]
  g0[3] = (unsigned)((ga >> 32) & 0x01FFFFFFu) | (2u << 30);  // addr hi + type=2
  i32x8 g1;
  g1[0] = (int)(1u << 16);                          // data_size=1 (2 bytes)
  g1[1] = (int)((tensor_w & 0xFFFFu) << 16);        // tensor_dim0[15:0] @ [63:48]
  g1[2] = (int)(((tensor_w >> 16) & 0xFFFFu) | ((tensor_h & 0xFFFFu) << 16));
  g1[3] = (int)(((tensor_h >> 16) & 0xFFFFu) | ((tile_w & 0xFFFFu) << 16));
  g1[4] = (int)(tile_h & 0xFFFFu);                  // tile_dim1 (tile_dim2=0)
  g1[5] = (int)stride_elems;                        // tensor_dim0_stride lo32
  g1[6] = 0;
  g1[7] = 0;
  i32x4 z4 = {0, 0, 0, 0};
#if defined(__clang_major__) && __clang_major__ >= 23
  i32x8 z8 = {0, 0, 0, 0, 0, 0, 0, 0};
  __builtin_amdgcn_tensor_load_to_lds(g0, g1, z4, z4, z8, 0);
#else
  __builtin_amdgcn_tensor_load_to_lds(g0, g1, z4, z4, 0);
#endif
}

__device__ __forceinline__ unsigned lds_off(const void* p) {
  return (unsigned)(size_t)p;  // LDS aperture: addr[31:0] is the LDS byte offset
}

// ---------------- Kernel 0: zero sumsq ----------------
__global__ void k_zero(float* p, int n) {
  int i = blockIdx.x * blockDim.x + threadIdx.x;
  if (i < n) p[i] = 0.f;
}

// ---------------- Pre-conversion: f32 -> bf16 (wo also folds rms_w) ----------
__global__ void k_cvt(const float* __restrict__ src, bf16* __restrict__ dst, int n) {
  int i = blockIdx.x * 256 + threadIdx.x;
  if (i < n) dst[i] = (bf16)src[i];
}
__global__ void k_cvt_scaled(const float* __restrict__ wo, const float* __restrict__ rmsw,
                             bf16* __restrict__ dst, int n) {
  int i = blockIdx.x * 256 + threadIdx.x;
  if (i < n) dst[i] = (bf16)(wo[i] * rmsw[i & (EE - 1)]);
}

// ---------------- Kernel 1: QKV projection + RoPE (TDM double-buffered) -----
// grid (M/128, H), block 256. Writes q,k,v bf16 in (B,H,S,D).
__global__ void __launch_bounds__(256) k_qkv_rope(
    const bf16* __restrict__ xbf, const float* __restrict__ fc,
    const bf16* __restrict__ wqb, const bf16* __restrict__ wkb,
    const bf16* __restrict__ wvb,
    bf16* __restrict__ qw, bf16* __restrict__ kw, bf16* __restrict__ vw) {
  __shared__ __align__(32) bf16 xs[2][128 * 32];
  __shared__ __align__(32) bf16 bsq[2][64 * 32];
  __shared__ __align__(32) bf16 bsk[2][64 * 32];
  __shared__ __align__(32) bf16 bsv[2][64 * 32];
  const int tid = threadIdx.x, lane = tid & 31, w = tid >> 5;
  const int m0 = blockIdx.x * 128;
  const int h  = blockIdx.y;
  v8f accq[4] = {}, acck[4] = {}, accv[4] = {};

  auto issue = [&](int buf, int kb) {
    if (w == 0)
      tdm_load_2d(lds_off(&xs[buf][0]), xbf + (size_t)m0 * EE + kb,
                  EE, BB * SS, 32, 128, EE);
    else if (w == 1)
      tdm_load_2d(lds_off(&bsq[buf][0]), wqb + (size_t)(h * 64) * EE + kb,
                  EE, EE, 32, 64, EE);
    else if (w == 2)
      tdm_load_2d(lds_off(&bsk[buf][0]), wkb + (size_t)(h * 64) * EE + kb,
                  EE, EE, 32, 64, EE);
    else if (w == 3)
      tdm_load_2d(lds_off(&bsv[buf][0]), wvb + (size_t)(h * 64) * EE + kb,
                  EE, EE, 32, 64, EE);
  };

  issue(0, 0);
  if (w < 4) __builtin_amdgcn_s_wait_tensorcnt(0);
  __syncthreads();
  int cur = 0;
  for (int kb = 0; kb < EE; kb += 32) {
    if (kb + 32 < EE) issue(cur ^ 1, kb + 32);  // prefetch next tiles via TDM
    v16bf a = frag_a(xs[cur], 32, w * 16, 0, lane);
#pragma unroll
    for (int nt = 0; nt < 4; ++nt) {
      accq[nt] = wmma_bf16(a, frag_b(bsq[cur], 32, nt * 16, 0, lane), accq[nt]);
      acck[nt] = wmma_bf16(a, frag_b(bsk[cur], 32, nt * 16, 0, lane), acck[nt]);
      accv[nt] = wmma_bf16(a, frag_b(bsv[cur], 32, nt * 16, 0, lane), accv[nt]);
    }
    if (kb + 32 < EE) {
      if (w < 4) __builtin_amdgcn_s_wait_tensorcnt(0);
      __syncthreads();
      cur ^= 1;
    }
  }
  // Epilogue: RoPE (pairs are adjacent lanes in C layout) + store (B,H,S,D)
  const int khalf = lane >> 4, nl = lane & 15;
#pragma unroll
  for (int r = 0; r < 8; ++r) {
    int grow = m0 + w * 16 + r + khalf * 8;
    int b = grow >> 13, s = grow & 8191;
    size_t ob = (((size_t)b * HH + h) * SS + s) * DD;
#pragma unroll
    for (int nt = 0; nt < 4; ++nt) {
      int d  = nt * 16 + nl;
      int fi = d >> 1;
      float c0 = fc[((size_t)s * 32 + fi) * 2];
      float s0 = fc[((size_t)s * 32 + fi) * 2 + 1];
      float vq = accq[nt][r];
      float pq = __shfl_xor(vq, 1, 32);
      qw[ob + d] = (bf16)((d & 1) ? (vq * c0 + pq * s0) : (vq * c0 - pq * s0));
      float vk = acck[nt][r];
      float pk = __shfl_xor(vk, 1, 32);
      kw[ob + d] = (bf16)((d & 1) ? (vk * c0 + pk * s0) : (vk * c0 - pk * s0));
      vw[ob + d] = (bf16)accv[nt][r];
    }
  }
}

// ---------------- Kernel 2: per-chunk KV = (k*k_dec)^T @ v ----------------
__global__ void __launch_bounds__(256) k_chunk_kv(
    const bf16* __restrict__ kw, const bf16* __restrict__ vw,
    float* __restrict__ kvs) {
  __shared__ __align__(32) bf16 s_kt[64 * 128];
  __shared__ __align__(32) bf16 s_vt[64 * 128];
  const int c = blockIdx.x, bh = blockIdx.y;
  const int h = bh & (HH - 1);
  const float slope = exp2f(-0.5f * (float)(h + 1));
  const int tid = threadIdx.x, lane = tid & 31, w = tid >> 5;
  const size_t base = ((size_t)bh * SS + (size_t)c * CC) * DD;
  for (int idx = tid; idx < CC * DD; idx += 256) {
    int j = idx >> 6, d = idx & 63;
    float kd = __expf(-slope * (float)(CC - 1 - j));
    s_kt[d * 128 + j] = (bf16)((float)kw[base + idx] * kd);
    s_vt[d * 128 + j] = vw[base + idx];
  }
  __syncthreads();
  const int mt = w & 3, ntb = (w >> 2) * 2;
  v8f acc[2] = {};
#pragma unroll
  for (int kb = 0; kb < 128; kb += 32) {
    v16bf a = frag_a(s_kt, 128, mt * 16, kb, lane);
    acc[0] = wmma_bf16(a, frag_b(s_vt, 128, (ntb + 0) * 16, kb, lane), acc[0]);
    acc[1] = wmma_bf16(a, frag_b(s_vt, 128, (ntb + 1) * 16, kb, lane), acc[1]);
  }
  float* outp = kvs + ((size_t)bh * NCC + c) * (DD * DD);
  const int khalf = lane >> 4, nl = lane & 15;
#pragma unroll
  for (int u = 0; u < 2; ++u)
#pragma unroll
    for (int r = 0; r < 8; ++r) {
      int d = mt * 16 + r + khalf * 8;
      int e = (ntb + u) * 16 + nl;
      outp[d * 64 + e] = acc[u][r];
    }
}

// ---------------- Kernel 3: in-place prefix scan of states ----------------
__global__ void __launch_bounds__(256) k_scan(float* __restrict__ kvs) {
  const int bh = blockIdx.x;
  const int h  = bh & (HH - 1);
  const float slope = exp2f(-0.5f * (float)(h + 1));
  const float cdec  = __expf(-slope * (float)CC);
  float st[16];
#pragma unroll
  for (int u = 0; u < 16; ++u) st[u] = 0.f;
  float* p = kvs + (size_t)bh * NCC * (DD * DD);
  for (int c = 0; c < NCC; ++c) {
    float* pc = p + (size_t)c * (DD * DD);
#pragma unroll
    for (int u = 0; u < 16; ++u) {
      int i = threadIdx.x + u * 256;
      float tmp = pc[i];
      pc[i] = st[u];
      st[u] = cdec * st[u] + tmp;
    }
  }
}

// ---------------- Kernel 4: per-chunk attention output ----------------
__global__ void __launch_bounds__(256) k_attn(
    const bf16* __restrict__ qw, const bf16* __restrict__ kw,
    const bf16* __restrict__ vw, const float* __restrict__ kvs,
    bf16* __restrict__ ow, float* __restrict__ ss) {
  __shared__ __align__(32) bf16 smem[128 * 128 + 64 * 128];  // 48 KB
  bf16* s_k  = smem;               // [128][64]   (dead after phase 1)
  bf16* s_st = smem + 128 * 64;    // [64][64]    S_pre^T (dead after phase 1)
  bf16* s_sc = smem;               // [128][128]  reuses k+st region
  bf16* s_vt = smem + 128 * 128;   // [64][128]   v^T
  const int c = blockIdx.x, bh = blockIdx.y;
  const int b = bh >> 4, h = bh & 15;
  const float slope = exp2f(-0.5f * (float)(h + 1));
  const int tid = threadIdx.x, lane = tid & 31, w = tid >> 5;
  const int khalf = lane >> 4, nl = lane & 15;
  const size_t base = ((size_t)bh * SS + (size_t)c * CC) * DD;
  // TDM for the natural-layout k tile; manual staging for transposed arrays
  if (w == 0) tdm_load_2d(lds_off(s_k), kw + base, DD, CC, DD, CC, DD);
  for (int idx = tid; idx < CC * DD; idx += 256) {
    int j = idx >> 6, d = idx & 63;
    s_vt[d * 128 + j] = vw[base + idx];
  }
  const float* stp = kvs + ((size_t)bh * NCC + c) * (DD * DD);
  for (int idx = tid; idx < DD * DD; idx += 256) {
    int d = idx >> 6, e = idx & 63;
    s_st[e * 64 + d] = (bf16)stp[idx];
  }
  // q fragments straight from global (private to this wave's 16-row strip)
  v16bf a0 = frag_a(qw + base, 64, w * 16, 0, lane);
  v16bf a1 = frag_a(qw + base, 64, w * 16, 32, lane);
  if (w == 0) __builtin_amdgcn_s_wait_tensorcnt(0);
  __syncthreads();
  // Phase 1: scores = q@k^T  and  inter = q@S_pre
  v8f accs[8] = {};
  v8f acci[4] = {};
#pragma unroll
  for (int nt = 0; nt < 8; ++nt) {
    accs[nt] = wmma_bf16(a0, frag_b(s_k, 64, nt * 16, 0, lane), accs[nt]);
    accs[nt] = wmma_bf16(a1, frag_b(s_k, 64, nt * 16, 32, lane), accs[nt]);
  }
#pragma unroll
  for (int nt = 0; nt < 4; ++nt) {
    acci[nt] = wmma_bf16(a0, frag_b(s_st, 64, nt * 16, 0, lane), acci[nt]);
    acci[nt] = wmma_bf16(a1, frag_b(s_st, 64, nt * 16, 32, lane), acci[nt]);
  }
  __syncthreads();  // everyone done with s_k / s_st; safe to overwrite as s_sc
  // Phase 2: mask+decay, C-layout -> A-layout via LDS (each wave owns its strip)
#pragma unroll
  for (int nt = 0; nt < 8; ++nt)
#pragma unroll
    for (int r = 0; r < 8; ++r) {
      int i = w * 16 + r + khalf * 8;
      int j = nt * 16 + nl;
      float sc = (i >= j) ? accs[nt][r] * __expf(-slope * (float)(i - j)) : 0.f;
      s_sc[i * 128 + j] = (bf16)sc;
    }
  // Phase 3: intra = scores @ v
  v8f acco[4] = {};
#pragma unroll
  for (int kb = 0; kb < 128; kb += 32) {
    v16bf a = frag_a(s_sc, 128, w * 16, kb, lane);
#pragma unroll
    for (int nt = 0; nt < 4; ++nt)
      acco[nt] = wmma_bf16(a, frag_b(s_vt, 128, nt * 16, kb, lane), acco[nt]);
  }
  // Epilogue: combine, store o (B,S,E) bf16, accumulate row sum-of-squares
#pragma unroll
  for (int r = 0; r < 8; ++r) {
    int i = w * 16 + r + khalf * 8;
    float qd = __expf(-slope * (float)(i + 1));
    size_t grow = (size_t)b * SS + (size_t)c * CC + i;
    float sq = 0.f;
#pragma unroll
    for (int nt = 0; nt < 4; ++nt) {
      float o = acco[nt][r] + qd * acci[nt][r];
      ow[grow * EE + h * 64 + nt * 16 + nl] = (bf16)o;
      sq += o * o;
    }
#pragma unroll
    for (int m = 1; m < 16; m <<= 1) sq += __shfl_xor(sq, m, 32);
    if (nl == 0) atomicAdd(&ss[grow], sq);
  }
}

// ---------------- Kernel 5: RMS-norm + output projection (TDM) ----------------
__global__ void __launch_bounds__(256) k_out_proj(
    const bf16* __restrict__ ow, const bf16* __restrict__ wob,
    const float* __restrict__ ss, float* __restrict__ out) {
  __shared__ __align__(32) bf16 as_[2][128 * 32];
  __shared__ __align__(32) bf16 bs[2][64 * 32];
  const int tid = threadIdx.x, lane = tid & 31, w = tid >> 5;
  const int m0 = blockIdx.x * 128, n0 = blockIdx.y * 64;
  v8f acc[4] = {};
  auto issue = [&](int buf, int kb) {
    if (w == 0)
      tdm_load_2d(lds_off(&as_[buf][0]), ow + (size_t)m0 * EE + kb,
                  EE, BB * SS, 32, 128, EE);
    else if (w == 1)
      tdm_load_2d(lds_off(&bs[buf][0]), wob + (size_t)n0 * EE + kb,
                  EE, EE, 32, 64, EE);
  };
  issue(0, 0);
  if (w < 2) __builtin_amdgcn_s_wait_tensorcnt(0);
  __syncthreads();
  int cur = 0;
  for (int kb = 0; kb < EE; kb += 32) {
    if (kb + 32 < EE) issue(cur ^ 1, kb + 32);
    v16bf a = frag_a(as_[cur], 32, w * 16, 0, lane);
#pragma unroll
    for (int nt = 0; nt < 4; ++nt)
      acc[nt] = wmma_bf16(a, frag_b(bs[cur], 32, nt * 16, 0, lane), acc[nt]);
    if (kb + 32 < EE) {
      if (w < 2) __builtin_amdgcn_s_wait_tensorcnt(0);
      __syncthreads();
      cur ^= 1;
    }
  }
  const int khalf = lane >> 4, nl = lane & 15;
#pragma unroll
  for (int r = 0; r < 8; ++r) {
    int grow = m0 + w * 16 + r + khalf * 8;
    float rs = rsqrtf(ss[grow] * (1.0f / (float)EE) + 1e-5f);
#pragma unroll
    for (int nt = 0; nt < 4; ++nt)
      out[(size_t)grow * EE + n0 + nt * 16 + nl] = acc[nt][r] * rs;
  }
}

// ---------------- Host launcher ----------------
extern "C" void kernel_launch(void* const* d_in, const int* in_sizes, int n_in,
                              void* d_out, int out_size, void* d_ws, size_t ws_size,
                              hipStream_t stream) {
  (void)in_sizes; (void)n_in; (void)out_size; (void)ws_size;
  const float* x    = (const float*)d_in[0];
  const float* fc   = (const float*)d_in[1];
  const float* wq   = (const float*)d_in[2];
  const float* wk   = (const float*)d_in[3];
  const float* wv   = (const float*)d_in[4];
  const float* wo   = (const float*)d_in[5];
  const float* rmsw = (const float*)d_in[6];
  float* out = (float*)d_out;

  const size_t nQ = (size_t)BB * HH * SS * DD;  // 16,777,216
  const size_t nX = (size_t)BB * SS * EE;       // 16,777,216
  const size_t nW = (size_t)EE * EE;            // 1,048,576
  char* ws = (char*)d_ws;
  size_t off = 0;
  bf16* qw  = (bf16*)(ws + off); off += nQ * sizeof(bf16);
  bf16* kw  = (bf16*)(ws + off); off += nQ * sizeof(bf16);
  bf16* vw  = (bf16*)(ws + off); off += nQ * sizeof(bf16);
  float* kvs = (float*)(ws + off); off += (size_t)BB * HH * NCC * DD * DD * sizeof(float);
  bf16* ow  = (bf16*)(ws + off); off += nX * sizeof(bf16);
  float* ss = (float*)(ws + off); off += (size_t)BB * SS * sizeof(float);
  bf16* xbf = (bf16*)(ws + off); off += nX * sizeof(bf16);
  bf16* wqb = (bf16*)(ws + off); off += nW * sizeof(bf16);
  bf16* wkb = (bf16*)(ws + off); off += nW * sizeof(bf16);
  bf16* wvb = (bf16*)(ws + off); off += nW * sizeof(bf16);
  bf16* wob = (bf16*)(ws + off);

  k_zero<<<(BB * SS + 255) / 256, 256, 0, stream>>>(ss, BB * SS);
  k_cvt<<<(int)((nX + 255) / 256), 256, 0, stream>>>(x, xbf, (int)nX);
  k_cvt<<<(int)((nW + 255) / 256), 256, 0, stream>>>(wq, wqb, (int)nW);
  k_cvt<<<(int)((nW + 255) / 256), 256, 0, stream>>>(wk, wkb, (int)nW);
  k_cvt<<<(int)((nW + 255) / 256), 256, 0, stream>>>(wv, wvb, (int)nW);
  k_cvt_scaled<<<(int)((nW + 255) / 256), 256, 0, stream>>>(wo, rmsw, wob, (int)nW);
  k_qkv_rope<<<dim3(BB * SS / 128, HH), 256, 0, stream>>>(xbf, fc, wqb, wkb, wvb,
                                                          qw, kw, vw);
  k_chunk_kv<<<dim3(NCC, BB * HH), 256, 0, stream>>>(kw, vw, kvs);
  k_scan<<<BB * HH, 256, 0, stream>>>(kvs);
  k_attn<<<dim3(NCC, BB * HH), 256, 0, stream>>>(qw, kw, vw, kvs, ow, ss);
  k_out_proj<<<dim3(BB * SS / 128, EE / 64), 256, 0, stream>>>(ow, wob, ss, out);
}
